// Gemma2MoeBlock_32925219291585
// MI455X (gfx1250) — compile-verified, matrix operand order checked by reading, not run on hardware
//
#include <hip/hip_runtime.h>
#include <hip/hip_bf16.h>
#include <stdint.h>

// Problem constants (fixed by the reference setup).
constexpr int T = 2048;   // B*S tokens
constexpr int H = 2048;   // hidden
constexpr int F = 4096;   // ffn
constexpr int E = 8;      // experts

typedef __attribute__((ext_vector_type(16))) __bf16 v16bf;
typedef __attribute__((ext_vector_type(8)))  float  v8f;

__device__ __forceinline__ unsigned int bf16_bits(float f) {
  unsigned int u = __builtin_bit_cast(unsigned int, f);
  return (u + 0x7fffu + ((u >> 16) & 1u)) >> 16;   // round-to-nearest-even
}
__device__ __forceinline__ unsigned int pack_bf16x2(float lo, float hi) {
  return (bf16_bits(lo) & 0xffffu) | (bf16_bits(hi) << 16);
}
__device__ __forceinline__ float gelu_tanh(float x) {
  const float c0 = 0.7978845608028654f;  // sqrt(2/pi)
  const float c1 = 0.044715f;
  float t = c0 * (x + c1 * x * x * x);
  float e = __expf(2.0f * t);
  return 0.5f * x * (1.0f + (e - 1.0f) / (e + 1.0f));
}

// ---------------------------------------------------------------------------
// Async global->LDS copy (CDNA5): no data VGPRs, tracked by ASYNCcnt.
// ---------------------------------------------------------------------------
#if __has_builtin(__builtin_amdgcn_global_load_async_to_lds_b128)
#define HAVE_ASYNC_LDS 1
typedef int v4i_vs __attribute__((vector_size(16)));
typedef __attribute__((address_space(1))) v4i_vs* gptr_v4i;
typedef __attribute__((address_space(3))) v4i_vs* lptr_v4i;
__device__ __forceinline__ void async_cp_b128(const void* g, void* l) {
  __builtin_amdgcn_global_load_async_to_lds_b128(
      (gptr_v4i)g, (lptr_v4i)l, 0, 0);
}
__device__ __forceinline__ void wait_async0() {
#if __has_builtin(__builtin_amdgcn_s_wait_asynccnt)
  __builtin_amdgcn_s_wait_asynccnt(0);
#else
  asm volatile("s_wait_asynccnt 0x0" ::: "memory");
#endif
}
#else
#define HAVE_ASYNC_LDS 0
__device__ __forceinline__ void wait_async0() {}
#endif

// ---------------------------------------------------------------------------
// Router: logits = x @ gate_w, softmax, top-2, renormalized combine weights.
// ---------------------------------------------------------------------------
__global__ void moe_router(const float* __restrict__ x,
                           const float* __restrict__ gw,
                           float* __restrict__ logits,
                           float* __restrict__ combine) {
  int t = blockIdx.x * blockDim.x + threadIdx.x;
  if (t >= T) return;
  float acc[E];
#pragma unroll
  for (int e = 0; e < E; ++e) acc[e] = 0.0f;
  const float* xr = x + (size_t)t * H;
  for (int h = 0; h < H; ++h) {
    float xv = xr[h];
#pragma unroll
    for (int e = 0; e < E; ++e) acc[e] += xv * gw[h * E + e];
  }
#pragma unroll
  for (int e = 0; e < E; ++e) logits[(size_t)t * E + e] = acc[e];

  float m = acc[0];
#pragma unroll
  for (int e = 1; e < E; ++e) m = fmaxf(m, acc[e]);
  float p[E];
#pragma unroll
  for (int e = 0; e < E; ++e) p[e] = __expf(acc[e] - m);
  int i1 = 0;
#pragma unroll
  for (int e = 1; e < E; ++e) if (p[e] > p[i1]) i1 = e;
  int i2 = (i1 == 0) ? 1 : 0;
#pragma unroll
  for (int e = 0; e < E; ++e) if (e != i1 && p[e] > p[i2]) i2 = e;
  float inv = 1.0f / (p[i1] + p[i2]);   // softmax norm cancels in the ratio
#pragma unroll
  for (int e = 0; e < E; ++e) combine[(size_t)t * E + e] = 0.0f;
  combine[(size_t)t * E + i1] = p[i1] * inv;
  combine[(size_t)t * E + i2] = p[i2] * inv;
}

// ---------------------------------------------------------------------------
// Elementwise fp32 -> bf16 (pairs).
// ---------------------------------------------------------------------------
__global__ void cvt_bf16(const float* __restrict__ in,
                         unsigned short* __restrict__ out, int npairs) {
  int i = blockIdx.x * blockDim.x + threadIdx.x;
  if (i >= npairs) return;
  const float2 v = reinterpret_cast<const float2*>(in)[i];
  reinterpret_cast<unsigned int*>(out)[i] = pack_bf16x2(v.x, v.y);
}

// ---------------------------------------------------------------------------
// Transpose + convert: in fp32 [K][N] row-major -> out bf16 [N][K] row-major.
// ---------------------------------------------------------------------------
__global__ __launch_bounds__(256) void transpose_cvt_bf16(
    const float* __restrict__ in, unsigned short* __restrict__ out,
    int K, int N) {
  __shared__ float tile[64][65];
  const int k0 = blockIdx.y * 64;
  const int n0 = blockIdx.x * 64;
#pragma unroll
  for (int it = 0; it < 16; ++it) {
    int i = threadIdx.x + it * 256;
    int r = i >> 6, c = i & 63;
    tile[r][c] = in[(size_t)(k0 + r) * N + n0 + c];
  }
  __syncthreads();
  unsigned int* outu = reinterpret_cast<unsigned int*>(out);
#pragma unroll
  for (int it = 0; it < 8; ++it) {
    int i = threadIdx.x + it * 256;
    int c = i >> 5, kp = i & 31;
    outu[(((size_t)(n0 + c) * K + k0) >> 1) + kp] =
        pack_bf16x2(tile[kp * 2][c], tile[kp * 2 + 1][c]);
  }
}

// ---------------------------------------------------------------------------
// WMMA fragment helpers.  LDS tiles: dword pairs of bf16, row stride 20 dwords
// (16B-aligned rows; the 16 lane rows land on disjoint 4-bank groups).
// A fragment (16x32 bf16, MxK): lanes 0-15 rows M=0..15 with K 0..7 / 16..23;
// lanes 16-31 same rows with K 8..15 / 24..31.  B (KxN) staged transposed
// [N][K] so the identical per-lane pattern applies with lane -> column.
// ---------------------------------------------------------------------------
union FragU { unsigned int u[8]; v16bf v; };

__device__ __forceinline__ void load_frag(FragU& f, const unsigned int* p) {
#pragma unroll
  for (int j = 0; j < 4; ++j) { f.u[j] = p[j]; f.u[4 + j] = p[8 + j]; }
}

#define LDS_STRIDE 20
#define TILE_U (128 * LDS_STRIDE)

// Copy one 16-byte chunk of a tile (global uint4 -> LDS), async if available.
__device__ __forceinline__ void tile_cp(unsigned int* ldsDst, const uint4* gSrc) {
#if HAVE_ASYNC_LDS
  async_cp_b128(gSrc, ldsDst);
#else
  *reinterpret_cast<uint4*>(ldsDst) = *gSrc;
#endif
}

// ---------------------------------------------------------------------------
// Pass A: h = gelu(x @ wg[e]) * (x @ wu[e]) -> bf16 scratch [T, F]
// Block tile 128x128, 8 waves as a 2(M) x 4(N) grid: each wave owns a 64x32
// sub-tile -> 4 A fragments reused across B columns, and each B-fragment
// load feeds 4 back-to-back WMMAs (amortizes the LDS wait).
// LDS ping-pong double buffering; one barrier per k-step.
// ---------------------------------------------------------------------------
__global__ __launch_bounds__(256) void moe_gate_up(
    const unsigned short* __restrict__ xb,    // [T][H]  bf16
    const unsigned short* __restrict__ wgT,   // [F][H]  bf16 (N-major)
    const unsigned short* __restrict__ wuT,   // [F][H]  bf16
    unsigned short* __restrict__ hbuf) {      // [T][F]  bf16
  __shared__ unsigned int As[2 * TILE_U];
  __shared__ unsigned int Bg[2 * TILE_U];
  __shared__ unsigned int Bu[2 * TILE_U];

  const int n0 = blockIdx.x * 128;
  const int m0 = blockIdx.y * 128;
  const int lane = threadIdx.x & 31;
  const int wv   = threadIdx.x >> 5;
  const int wvM  = wv >> 2;                 // 0..1  (M tile of 64)
  const int wvN  = wv & 3;                  // 0..3  (N tile of 32)
  const int half = lane >> 4;
  const int l    = lane & 15;

  const uint4* A4 = reinterpret_cast<const uint4*>(xb);
  const uint4* G4 = reinterpret_cast<const uint4*>(wgT);
  const uint4* U4 = reinterpret_cast<const uint4*>(wuT);

  v8f accg[4][2], accu[4][2];
#pragma unroll
  for (int fa = 0; fa < 4; ++fa)
#pragma unroll
    for (int nt = 0; nt < 2; ++nt) { accg[fa][nt] = (v8f){}; accu[fa][nt] = (v8f){}; }

  // Tile copy geometry: 128 rows x 32 bf16 = 512 uint4; 2 per thread.
  const int r0 = threadIdx.x >> 2;          // 0..63
  const int q0 = threadIdx.x & 3;           // uint4 slot in row

  auto stage = [&](int kkk, int p) {
#pragma unroll
    for (int it = 0; it < 2; ++it) {
      int r = r0 + it * 64;
      unsigned int lo = p * TILE_U + r * LDS_STRIDE + q0 * 4;
      size_t ga = (((size_t)(m0 + r) * H + kkk) >> 3) + q0;
      size_t gw_ = (((size_t)(n0 + r) * H + kkk) >> 3) + q0;
      tile_cp(&As[lo], &A4[ga]);
      tile_cp(&Bg[lo], &G4[gw_]);
      tile_cp(&Bu[lo], &U4[gw_]);
    }
  };

  const int fragOff = l * LDS_STRIDE + half * 4;   // + row_base*LDS_STRIDE

  stage(0, 0);
  int p = 0;
  for (int kk = 0; kk < H; kk += 32, p ^= 1) {
    wait_async0();
    __syncthreads();
    const int kn = kk + 32;
    if (kn < H) {
      stage(kn, p ^ 1);
      if (kn + 32 < H) {   // L2 prefetch two tiles ahead
        __builtin_prefetch(&A4[(((size_t)(m0 + r0) * H + kn + 32) >> 3) + q0], 0, 1);
        __builtin_prefetch(&G4[(((size_t)(n0 + r0) * H + kn + 32) >> 3) + q0], 0, 1);
        __builtin_prefetch(&U4[(((size_t)(n0 + r0) * H + kn + 32) >> 3) + q0], 0, 1);
      }
    }

    const unsigned int* Asb = &As[p * TILE_U];
    const unsigned int* Bgb = &Bg[p * TILE_U];
    const unsigned int* Bub = &Bu[p * TILE_U];

    FragU a[4];
#pragma unroll
    for (int fa = 0; fa < 4; ++fa)
      load_frag(a[fa], &Asb[(wvM * 64 + fa * 16) * LDS_STRIDE + fragOff]);

#pragma unroll
    for (int nt = 0; nt < 2; ++nt) {
      FragU bg, bu;
      load_frag(bg, &Bgb[(wvN * 32 + nt * 16) * LDS_STRIDE + fragOff]);
      load_frag(bu, &Bub[(wvN * 32 + nt * 16) * LDS_STRIDE + fragOff]);
#pragma unroll
      for (int fa = 0; fa < 4; ++fa)
        accg[fa][nt] = __builtin_amdgcn_wmma_f32_16x16x32_bf16(
            false, a[fa].v, false, bg.v, (short)0, accg[fa][nt], false, false);
#pragma unroll
      for (int fa = 0; fa < 4; ++fa)
        accu[fa][nt] = __builtin_amdgcn_wmma_f32_16x16x32_bf16(
            false, a[fa].v, false, bu.v, (short)0, accu[fa][nt], false, false);
    }
  }

  // Epilogue: gelu(g) * u -> bf16 h scratch.
  // C layout: VGPR i -> M = i (lanes 0-15) / 8+i (lanes 16-31), N = l.
#pragma unroll
  for (int fa = 0; fa < 4; ++fa) {
#pragma unroll
    for (int nt = 0; nt < 2; ++nt) {
#pragma unroll
      for (int i = 0; i < 8; ++i) {
        float hv = gelu_tanh(accg[fa][nt][i]) * accu[fa][nt][i];
        int row = m0 + wvM * 64 + fa * 16 + i + half * 8;
        int col = n0 + wvN * 32 + nt * 16 + l;
        hbuf[(size_t)row * F + col] = (unsigned short)bf16_bits(hv);
      }
    }
  }
}

// ---------------------------------------------------------------------------
// Pass B: out (+)= (h @ wd[e]) * combine[:, e]
// Same 2x4 wave grid: 4 A fragments x 2 B columns per wave.
// ---------------------------------------------------------------------------
__global__ __launch_bounds__(256) void moe_down(
    const unsigned short* __restrict__ hbuf,  // [T][F] bf16
    const unsigned short* __restrict__ wdT,   // [H][F] bf16 (N-major)
    const float* __restrict__ combine,
    float* __restrict__ out,
    int e, int init) {
  __shared__ unsigned int As[2 * TILE_U];
  __shared__ unsigned int Bt[2 * TILE_U];
  __shared__ float cw[128];

  const int n0 = blockIdx.x * 128;
  const int m0 = blockIdx.y * 128;
  const int lane = threadIdx.x & 31;
  const int wv   = threadIdx.x >> 5;
  const int wvM  = wv >> 2;
  const int wvN  = wv & 3;
  const int half = lane >> 4;
  const int l    = lane & 15;

  if (threadIdx.x < 128)
    cw[threadIdx.x] = combine[(size_t)(m0 + threadIdx.x) * E + e];

  const uint4* A4 = reinterpret_cast<const uint4*>(hbuf);
  const uint4* B4 = reinterpret_cast<const uint4*>(wdT);

  v8f acc[4][2];
#pragma unroll
  for (int fa = 0; fa < 4; ++fa)
#pragma unroll
    for (int nt = 0; nt < 2; ++nt) acc[fa][nt] = (v8f){};

  const int r0 = threadIdx.x >> 2;
  const int q0 = threadIdx.x & 3;

  auto stage = [&](int kkk, int p) {
#pragma unroll
    for (int it = 0; it < 2; ++it) {
      int r = r0 + it * 64;
      unsigned int lo = p * TILE_U + r * LDS_STRIDE + q0 * 4;
      tile_cp(&As[lo], &A4[(((size_t)(m0 + r) * F + kkk) >> 3) + q0]);
      tile_cp(&Bt[lo], &B4[(((size_t)(n0 + r) * F + kkk) >> 3) + q0]);
    }
  };

  const int fragOff = l * LDS_STRIDE + half * 4;

  stage(0, 0);
  int p = 0;
  for (int kk = 0; kk < F; kk += 32, p ^= 1) {
    wait_async0();
    __syncthreads();
    const int kn = kk + 32;
    if (kn < F) {
      stage(kn, p ^ 1);
      if (kn + 32 < F) {
        __builtin_prefetch(&A4[(((size_t)(m0 + r0) * F + kn + 32) >> 3) + q0], 0, 1);
        __builtin_prefetch(&B4[(((size_t)(n0 + r0) * F + kn + 32) >> 3) + q0], 0, 1);
      }
    }

    const unsigned int* Asb = &As[p * TILE_U];
    const unsigned int* Btb = &Bt[p * TILE_U];

    FragU a[4];
#pragma unroll
    for (int fa = 0; fa < 4; ++fa)
      load_frag(a[fa], &Asb[(wvM * 64 + fa * 16) * LDS_STRIDE + fragOff]);

#pragma unroll
    for (int nt = 0; nt < 2; ++nt) {
      FragU b;
      load_frag(b, &Btb[(wvN * 32 + nt * 16) * LDS_STRIDE + fragOff]);
#pragma unroll
      for (int fa = 0; fa < 4; ++fa)
        acc[fa][nt] = __builtin_amdgcn_wmma_f32_16x16x32_bf16(
            false, a[fa].v, false, b.v, (short)0, acc[fa][nt], false, false);
    }
  }

#pragma unroll
  for (int fa = 0; fa < 4; ++fa) {
#pragma unroll
    for (int nt = 0; nt < 2; ++nt) {
#pragma unroll
      for (int i = 0; i < 8; ++i) {
        int rl = wvM * 64 + fa * 16 + i + half * 8;
        float v = acc[fa][nt][i] * cw[rl];
        size_t idx = (size_t)(m0 + rl) * H + n0 + wvN * 32 + nt * 16 + l;
        if (init) out[idx] = v;
        else      out[idx] += v;
      }
    }
  }
}

// ---------------------------------------------------------------------------
// Launch. Workspace layout (all re-written every call; deterministic):
//   combine  : T*E f32            (64 KB)
//   xb       : T*H bf16           (8 MB)
//   hbuf     : T*F bf16           (16 MB)
//   wgT/wuT  : F*H bf16 each      (16 MB x2, per-expert reuse)
//   wdT      : H*F bf16           (16 MB, per-expert reuse)
// out = [T*H floats] ++ [T*E logits].
// ---------------------------------------------------------------------------
extern "C" void kernel_launch(void* const* d_in, const int* in_sizes, int n_in,
                              void* d_out, int out_size, void* d_ws, size_t ws_size,
                              hipStream_t stream) {
  const float* x      = (const float*)d_in[0];
  const float* gate_w = (const float*)d_in[1];
  const float* wg     = (const float*)d_in[2];
  const float* wu     = (const float*)d_in[3];
  const float* wd     = (const float*)d_in[4];

  float* out    = (float*)d_out;
  float* logits = out + (size_t)T * H;

  char* ws = (char*)d_ws;
  float*          combine = (float*)ws;                 ws += 1 << 16;
  unsigned short* xb      = (unsigned short*)ws;        ws += (size_t)T * H * 2;
  unsigned short* hbuf    = (unsigned short*)ws;        ws += (size_t)T * F * 2;
  unsigned short* wgT     = (unsigned short*)ws;        ws += (size_t)F * H * 2;
  unsigned short* wuT     = (unsigned short*)ws;        ws += (size_t)F * H * 2;
  unsigned short* wdT     = (unsigned short*)ws;

  moe_router<<<T / 256, 256, 0, stream>>>(x, gate_w, logits, combine);
  cvt_bf16<<<(T * H / 2) / 256, 256, 0, stream>>>(x, xb, T * H / 2);

  for (int e = 0; e < E; ++e) {
    // wg/wu: [H][F] -> [F][H];  wd: [F][H] -> [H][F]
    transpose_cvt_bf16<<<dim3(F / 64, H / 64), 256, 0, stream>>>(
        wg + (size_t)e * H * F, wgT, H, F);
    transpose_cvt_bf16<<<dim3(F / 64, H / 64), 256, 0, stream>>>(
        wu + (size_t)e * H * F, wuT, H, F);
    transpose_cvt_bf16<<<dim3(H / 64, F / 64), 256, 0, stream>>>(
        wd + (size_t)e * F * H, wdT, F, H);

    moe_gate_up<<<dim3(F / 128, T / 128), 256, 0, stream>>>(xb, wgT, wuT, hbuf);
    moe_down<<<dim3(H / 128, T / 128), 256, 0, stream>>>(
        hbuf, wdT, combine, out, e, e == 0 ? 1 : 0);
  }
}
